// MultiHeadAttention_56684978372816
// MI455X (gfx1250) — matrix operand order repaired and checked by reading, not compiled, not run on hardware
//
#include <hip/hip_runtime.h>
#include <hip/hip_bf16.h>

typedef __attribute__((ext_vector_type(16))) _Float16 v16h;
typedef __attribute__((ext_vector_type(8)))  _Float16 v8h;
typedef __attribute__((ext_vector_type(4)))  _Float16 v4h;
typedef __attribute__((ext_vector_type(8)))  float    v8f;
typedef __attribute__((ext_vector_type(4)))  float    f32x4;
typedef __attribute__((ext_vector_type(4)))  unsigned int u32x4;
typedef __attribute__((ext_vector_type(8)))  int      i32x8;
typedef __attribute__((ext_vector_type(4)))  int      i32x4;

#define D_MODEL  2048
#define N_HEADS  16
#define HEAD_DIM 128
#define SEQ      2048
#define BATCH    2
#define ROWS     (BATCH * SEQ)   // 4096
#define KT       64              // GEMM k-tile
#define NIT      (D_MODEL / KT)  // 32 pipeline stages
#define ASTRIDE  72              // 64 halves + 4dw TDM pad (144B, 16B-aligned)

#if __has_builtin(__builtin_amdgcn_tensor_load_to_lds)
#define HAVE_TDM 1
#endif

// ---------------- helpers ----------------
__device__ __forceinline__ v16h ld_frag(const _Float16* p0, const _Float16* p1) {
  union { v16h v; u32x4 q[2]; } u;
  u.q[0] = *(const u32x4*)p0;
  u.q[1] = *(const u32x4*)p1;
  return u.v;
}

__device__ __forceinline__ v8f wmma_f16(v16h a, v16h b, v8f c) {
  return __builtin_amdgcn_wmma_f32_16x16x32_f16(
      false, a, false, b, (short)0, c, false, false);
}

__device__ __forceinline__ float redmax16(float v) {
  v = fmaxf(v, __shfl_xor(v, 1, 32));
  v = fmaxf(v, __shfl_xor(v, 2, 32));
  v = fmaxf(v, __shfl_xor(v, 4, 32));
  v = fmaxf(v, __shfl_xor(v, 8, 32));
  return v;
}
__device__ __forceinline__ float redsum16(float v) {
  v += __shfl_xor(v, 1, 32);
  v += __shfl_xor(v, 2, 32);
  v += __shfl_xor(v, 4, 32);
  v += __shfl_xor(v, 8, 32);
  return v;
}

// ---------------- TDM descriptor builders ----------------
__device__ __forceinline__ u32x4 tdm_g0(unsigned lds_addr, unsigned long long ga) {
  u32x4 g;
  g.x = 1u;                                    // count=1, user mode
  g.y = lds_addr;                              // LDS byte address
  g.z = (unsigned)(ga & 0xFFFFFFFFull);        // global_addr[31:0]
  g.w = (unsigned)((ga >> 32) & 0x1FFFFFFull)  // global_addr[56:32]
        | (2u << 30);                          // type = 2
  return g;
}
// 2D tensor, data_size=2B, LDS row padding: interval 2^(padI+1) dw, amount padA+1 dw
__device__ __forceinline__ i32x8 tdm_g1(unsigned dim0, unsigned dim1,
                                        unsigned tile0, unsigned tile1,
                                        unsigned stride0,
                                        unsigned padI, unsigned padA) {
  i32x8 g;
  g[0] = (int)((1u << 16) | (1u << 20) | (padI << 22) | (padA << 25));
  g[1] = (int)((dim0 & 0xFFFFu) << 16);
  g[2] = (int)(((dim0 >> 16) & 0xFFFFu) | ((dim1 & 0xFFFFu) << 16));
  g[3] = (int)(((dim1 >> 16) & 0xFFFFu) | ((tile0 & 0xFFFFu) << 16));
  g[4] = (int)(tile1 & 0xFFFFu);
  g[5] = (int)stride0;
  g[6] = 0;
  g[7] = 0;
  return g;
}

__device__ __forceinline__ void tdm_load(u32x4 g0, i32x8 g1) {
#ifdef HAVE_TDM
  const i32x4 z4 = {0, 0, 0, 0};
#if __clang_major__ >= 23
  const i32x8 z8 = {0, 0, 0, 0, 0, 0, 0, 0};
  __builtin_amdgcn_tensor_load_to_lds(g0, g1, z4, z4, z8, 0);
#else
  __builtin_amdgcn_tensor_load_to_lds(g0, g1, z4, z4, 0);
#endif
#endif
}

__device__ __forceinline__ unsigned lds_addr_of(const void* p) {
  return (unsigned)(unsigned long long)(size_t)p;  // low 32 bits = LDS offset
}

// ---------------- prep: f32 -> f16 ----------------
__global__ void cvt_f32_to_f16(const float* __restrict__ in,
                               _Float16* __restrict__ out, int n4) {
  int i = blockIdx.x * blockDim.x + threadIdx.x;
  if (i < n4) {
    f32x4 v = ((const f32x4*)in)[i];
    v4h h;
    h.x = (_Float16)v.x; h.y = (_Float16)v.y;
    h.z = (_Float16)v.z; h.w = (_Float16)v.w;
    ((v4h*)out)[i] = h;
  }
}

// ---------------- prep: W [k][n] f32 -> Wt [n][k] f16 (x3 mats) ----------------
__global__ void transpose_cvt(const float* __restrict__ Wq,
                              const float* __restrict__ Wk,
                              const float* __restrict__ Wv,
                              _Float16* __restrict__ Wt) {
  __shared__ float tile[32][33];
  const int mat = blockIdx.z;
  const float* src = (mat == 0) ? Wq : ((mat == 1) ? Wk : Wv);
  _Float16* dst = Wt + (size_t)mat * D_MODEL * D_MODEL;
  const int n0 = blockIdx.x * 32, k0 = blockIdx.y * 32;
  const int tx = threadIdx.x, ty = threadIdx.y;   // 32 x 8
#pragma unroll
  for (int r = 0; r < 4; ++r)
    tile[ty + 8 * r][tx] = src[(size_t)(k0 + ty + 8 * r) * D_MODEL + n0 + tx];
  __syncthreads();
#pragma unroll
  for (int r = 0; r < 4; ++r)
    dst[(size_t)(n0 + ty + 8 * r) * D_MODEL + k0 + tx] = (_Float16)tile[tx][ty + 8 * r];
}

// ---------------- QKV GEMM: double-buffered TDM pipeline ----------------
__global__ __launch_bounds__(256) void qkv_gemm(const _Float16* __restrict__ Xh,
                                                const _Float16* __restrict__ Wt,
                                                _Float16* __restrict__ QKV) {
  __shared__ _Float16 Asm[2][128 * ASTRIDE];   // [m][k] tiles, ping-pong
  __shared__ _Float16 Bsm[2][128 * ASTRIDE];   // [n][k] tiles, ping-pong

  const int mat = blockIdx.z;
  const _Float16* Wtm = Wt + (size_t)mat * D_MODEL * D_MODEL;
  _Float16* Om = QKV + (size_t)mat * ROWS * D_MODEL;

  const int tid = threadIdx.x;
  const int lane = tid & 31, wv = tid >> 5;
  const int lo = lane & 15, hi = lane >> 4;
  const int wm = wv & 3, wn = wv >> 2;            // 4 x 2 wave grid
  const int rowBase = blockIdx.y * 128, colBase = blockIdx.x * 128;

  const v8f zero = {0, 0, 0, 0, 0, 0, 0, 0};
  v8f acc[2][4];
#pragma unroll
  for (int mt = 0; mt < 2; ++mt)
#pragma unroll
    for (int nt = 0; nt < 4; ++nt) acc[mt][nt] = zero;

  const _Float16* aBase = Xh  + (size_t)rowBase * D_MODEL;
  const _Float16* bBase = Wtm + (size_t)colBase * D_MODEL;

#ifdef HAVE_TDM
  // tile = KT x 128 rows; row = 32 dw, pad 4 dw -> ASTRIDE halves
  const i32x8 g1A = tdm_g1(D_MODEL, ROWS,    KT, 128, D_MODEL, 4, 3);
  const i32x8 g1B = tdm_g1(D_MODEL, D_MODEL, KT, 128, D_MODEL, 4, 3);
  const unsigned ldsA[2] = {lds_addr_of(Asm[0]), lds_addr_of(Asm[1])};
  const unsigned ldsB[2] = {lds_addr_of(Bsm[0]), lds_addr_of(Bsm[1])};
  if (tid < 32) {  // wave 0 drives the Tensor Data Mover: stage 0 into buf 0
    tdm_load(tdm_g0(ldsA[0], (unsigned long long)(size_t)aBase), g1A);
    tdm_load(tdm_g0(ldsB[0], (unsigned long long)(size_t)bBase), g1B);
  }
#else
  const int ldRow = tid >> 1;
  const int ldChunk = (tid & 1) * 32;    // halves
#endif

  for (int it = 0; it < NIT; ++it) {
    const int cur = it & 1;
    __syncthreads();   // buf[cur^1] reads from stage it-1 complete
#ifdef HAVE_TDM
    if (tid < 32) {
      if (it + 1 < NIT) {    // issue stage it+1 into the other buffer
        const _Float16* an = aBase + (it + 1) * KT;
        const _Float16* bn = bBase + (it + 1) * KT;
        tdm_load(tdm_g0(ldsA[cur ^ 1], (unsigned long long)(size_t)an), g1A);
        tdm_load(tdm_g0(ldsB[cur ^ 1], (unsigned long long)(size_t)bn), g1B);
        __builtin_amdgcn_s_wait_tensorcnt(2);   // stage it landed
      } else {
        __builtin_amdgcn_s_wait_tensorcnt(0);   // final stage landed
      }
    }
#else
    {
      const _Float16* aSrc = aBase + (size_t)ldRow * D_MODEL + it * KT + ldChunk;
      const _Float16* bSrc = bBase + (size_t)ldRow * D_MODEL + it * KT + ldChunk;
      _Float16* aDst = Asm[cur] + ldRow * ASTRIDE + ldChunk;
      _Float16* bDst = Bsm[cur] + ldRow * ASTRIDE + ldChunk;
#pragma unroll
      for (int c = 0; c < 4; ++c) {
        *(u32x4*)(aDst + c * 8) = *(const u32x4*)(aSrc + c * 8);
        *(u32x4*)(bDst + c * 8) = *(const u32x4*)(bSrc + c * 8);
      }
    }
#endif
    __syncthreads();   // buf[cur] published

    const _Float16* As = Asm[cur];
    const _Float16* Bs = Bsm[cur];
#pragma unroll
    for (int ks = 0; ks < 2; ++ks) {       // two 32-K WMMA substeps per stage
      v16h af[2], bf[4];
#pragma unroll
      for (int mt = 0; mt < 2; ++mt) {
        const int r = wm * 32 + mt * 16 + lo;
        af[mt] = ld_frag(&As[r * ASTRIDE + ks * 32 + hi * 8],
                         &As[r * ASTRIDE + ks * 32 + hi * 8 + 16]);
      }
#pragma unroll
      for (int nt = 0; nt < 4; ++nt) {
        const int c = wn * 64 + nt * 16 + lo;
        bf[nt] = ld_frag(&Bs[c * ASTRIDE + ks * 32 + hi * 8],
                         &Bs[c * ASTRIDE + ks * 32 + hi * 8 + 16]);
      }
#pragma unroll
      for (int mt = 0; mt < 2; ++mt)
#pragma unroll
        for (int nt = 0; nt < 4; ++nt)
          acc[mt][nt] = wmma_f16(af[mt], bf[nt], acc[mt][nt]);
    }
  }

  const float qscale = (mat == 0) ? 0.08838834764831845f : 1.0f;  // 1/sqrt(128)
#pragma unroll
  for (int mt = 0; mt < 2; ++mt)
#pragma unroll
    for (int nt = 0; nt < 4; ++nt)
#pragma unroll
      for (int i = 0; i < 8; ++i) {
        const int r = rowBase + wm * 32 + mt * 16 + i + 8 * hi;
        const int c = colBase + wn * 64 + nt * 16 + lo;
        const int b = r >> 11, sIn = r & 2047;
        const int h = c >> 7,  d   = c & 127;
        Om[((size_t)(b * N_HEADS + h) * SEQ + sIn) * HEAD_DIM + d] =
            (_Float16)(acc[mt][nt][i] * qscale);
      }
}

// ---------------- causal flash attention: double-buffered K (TDM) + V ----------------
__global__ __launch_bounds__(128) void attn_fwd(const _Float16* __restrict__ Qh,
                                                const _Float16* __restrict__ Kh,
                                                const _Float16* __restrict__ Vh,
                                                float* __restrict__ Out) {
  __shared__ _Float16 Ksm[2][32 * 136];    // [key][d], 136-half stride (TDM pad)
  __shared__ _Float16 Vsm[2][128 * 40];    // [d][key] transposed
  __shared__ _Float16 Psm[4 * 16 * 40];    // per-wave P tile

  const int bh = blockIdx.y;
  const int b = bh >> 4, h = bh & 15;
  const int qBase = blockIdx.x * 64;

  const int tid = threadIdx.x;
  const int lane = tid & 31, wv = tid >> 5;
  const int lo = lane & 15, hi = lane >> 4;

  const _Float16* Qrow =
      Qh + ((size_t)bh * SEQ + qBase + wv * 16 + lo) * HEAD_DIM;
  v16h qf[4];
#pragma unroll
  for (int f = 0; f < 4; ++f) {
    const int off = f * 32 + hi * 8;
    qf[f] = ld_frag(Qrow + off, Qrow + off + 16);
  }

  const v8f zero = {0, 0, 0, 0, 0, 0, 0, 0};
  v8f accO[8];
#pragma unroll
  for (int n = 0; n < 8; ++n) accO[n] = zero;
  float mrun[8], lrun[8];
#pragma unroll
  for (int i = 0; i < 8; ++i) { mrun[i] = -1e30f; lrun[i] = 0.0f; }

  const int ldRow = tid >> 2;          // 0..31 (key within block)
  const int ldCol = (tid & 3) * 32;    // d offset (halves)
  const _Float16* Ksrc = Kh + (size_t)bh * SEQ * HEAD_DIM;
  const _Float16* Vsrc = Vh + (size_t)bh * SEQ * HEAD_DIM;

  const int qRow0 = qBase + wv * 16 + 8 * hi;
  const int nKB = (qBase >> 5) + 2;    // >= 2 always

#ifdef HAVE_TDM
  // tile = 128 (d) x 32 (keys); row 64 dw, pad 4 dw -> 136-half stride
  const i32x8 g1K = tdm_g1(HEAD_DIM, SEQ, HEAD_DIM, 32, HEAD_DIM, 5, 3);
  const unsigned ldsK[2] = {lds_addr_of(Ksm[0]), lds_addr_of(Ksm[1])};
  if (tid < 32)   // stage K block 0 into buffer 0
    tdm_load(tdm_g0(ldsK[0], (unsigned long long)(size_t)Ksrc), g1K);
#endif
  // stage V block 0 into registers
  v8h vreg[4];
  {
    const _Float16* vp = Vsrc + (size_t)ldRow * HEAD_DIM + ldCol;
#pragma unroll
    for (int c = 0; c < 4; ++c) vreg[c] = *(const v8h*)(vp + c * 8);
  }

  for (int j = 0; j < nKB; ++j) {
    const int cur = j & 1;
    const int kBase = j * 32;
    __syncthreads();   // buf[cur^1] reads from block j-1 complete
#ifdef HAVE_TDM
    if (tid < 32) {
      if (j + 1 < nKB) {
        tdm_load(tdm_g0(ldsK[cur ^ 1],
                        (unsigned long long)(size_t)(Ksrc + (size_t)(kBase + 32) * HEAD_DIM)),
                 g1K);
        __builtin_amdgcn_s_wait_tensorcnt(1);   // block j landed
      } else {
        __builtin_amdgcn_s_wait_tensorcnt(0);
      }
    }
#else
    {
      const _Float16* kp = Ksrc + (size_t)(kBase + ldRow) * HEAD_DIM + ldCol;
      _Float16* kd = Ksm[cur] + ldRow * 136 + ldCol;
#pragma unroll
      for (int c = 0; c < 4; ++c)
        *(u32x4*)(kd + c * 8) = *(const u32x4*)(kp + c * 8);
    }
#endif
    // V(j): registers -> LDS transposed [d][key]
#pragma unroll
    for (int c = 0; c < 4; ++c)
#pragma unroll
      for (int e = 0; e < 8; ++e)
        Vsm[cur][(ldCol + c * 8 + e) * 40 + ldRow] = vreg[c][e];
    // prefetch V(j+1) into registers
    if (j + 1 < nKB) {
      const _Float16* vp = Vsrc + (size_t)(kBase + 32 + ldRow) * HEAD_DIM + ldCol;
#pragma unroll
      for (int c = 0; c < 4; ++c) vreg[c] = *(const v8h*)(vp + c * 8);
      __builtin_prefetch(vp + 32 * HEAD_DIM, 0, 1);   // V(j+2) -> cache
    }
    __syncthreads();   // K[cur] (TDM) + V[cur] published

    const _Float16* Ks = Ksm[cur];
    const _Float16* Vs = Vsm[cur];

    // --- S = Q * K^T over 32 keys ---
    v8f s0 = zero, s1 = zero;
#pragma unroll
    for (int kd = 0; kd < 4; ++kd) {
      const v16h b0 = ld_frag(&Ks[lo * 136 + kd * 32 + hi * 8],
                              &Ks[lo * 136 + kd * 32 + hi * 8 + 16]);
      const v16h b1 = ld_frag(&Ks[(16 + lo) * 136 + kd * 32 + hi * 8],
                              &Ks[(16 + lo) * 136 + kd * 32 + hi * 8 + 16]);
      s0 = wmma_f16(qf[kd], b0, s0);
      s1 = wmma_f16(qf[kd], b1, s1);
    }

    // --- causal mask + online softmax ---
    _Float16* pRow = Psm + wv * 640;
#pragma unroll
    for (int i = 0; i < 8; ++i) {
      const int q = qRow0 + i;
      const int kc = kBase + lo;
      float x0 = (kc <= q)      ? s0[i] : -1e6f;
      float x1 = (kc + 16 <= q) ? s1[i] : -1e6f;
      float bm = redmax16(fmaxf(x0, x1));
      const float mnew = fmaxf(mrun[i], bm);
      const float p0 = __expf(x0 - mnew);
      const float p1 = __expf(x1 - mnew);
      const float rs = redsum16(p0 + p1);
      const float sc = __expf(mrun[i] - mnew);
      lrun[i] = lrun[i] * sc + rs;
      mrun[i] = mnew;
#pragma unroll
      for (int n = 0; n < 8; ++n) accO[n][i] *= sc;
      _Float16* pp = pRow + (i + 8 * hi) * 40;
      pp[lo]      = (_Float16)p0;
      pp[lo + 16] = (_Float16)p1;
    }

    // --- O += P * V ---
    const v16h pf = ld_frag(pRow + lo * 40 + hi * 8,
                            pRow + lo * 40 + hi * 8 + 16);
#pragma unroll
    for (int n = 0; n < 8; ++n) {
      const v16h vf = ld_frag(&Vs[(n * 16 + lo) * 40 + hi * 8],
                              &Vs[(n * 16 + lo) * 40 + hi * 8 + 16]);
      accO[n] = wmma_f16(pf, vf, accO[n]);
    }
  }

  // --- epilogue: normalize and store [b][s][h*128+d] f32 ---
#pragma unroll
  for (int i = 0; i < 8; ++i) {
    const float inv = 1.0f / lrun[i];
    const int q = qRow0 + i;
    float* dst = Out + ((size_t)b * SEQ + q) * D_MODEL + h * HEAD_DIM;
#pragma unroll
    for (int n = 0; n < 8; ++n)
      dst[n * 16 + lo] = accO[n][i] * inv;
  }
}

// ---------------- launcher ----------------
extern "C" void kernel_launch(void* const* d_in, const int* in_sizes, int n_in,
                              void* d_out, int out_size, void* d_ws, size_t ws_size,
                              hipStream_t stream) {
  (void)in_sizes; (void)n_in; (void)out_size; (void)ws_size;
  const float* x  = (const float*)d_in[0];
  const float* Wq = (const float*)d_in[1];
  const float* Wk = (const float*)d_in[2];
  const float* Wv = (const float*)d_in[3];
  float* out = (float*)d_out;

  char* ws = (char*)d_ws;
  const size_t XH_BYTES  = (size_t)ROWS * D_MODEL * sizeof(_Float16);
  const size_t WT_BYTES  = (size_t)D_MODEL * D_MODEL * sizeof(_Float16);
  const size_t QKV_BYTES = (size_t)BATCH * N_HEADS * SEQ * HEAD_DIM * sizeof(_Float16);

  _Float16* Xh  = (_Float16*)(ws);
  _Float16* Wt  = (_Float16*)(ws + XH_BYTES);
  _Float16* QKV = (_Float16*)(ws + XH_BYTES + 3 * WT_BYTES);
  _Float16* Qh = QKV;
  _Float16* Kh = QKV + QKV_BYTES / sizeof(_Float16);
  _Float16* Vh = QKV + 2 * (QKV_BYTES / sizeof(_Float16));

  {
    const int n4 = ROWS * D_MODEL / 4;
    cvt_f32_to_f16<<<n4 / 256, 256, 0, stream>>>(x, Xh, n4);
  }
  transpose_cvt<<<dim3(D_MODEL / 32, D_MODEL / 32, 3), dim3(32, 8), 0, stream>>>(
      Wq, Wk, Wv, Wt);
  qkv_gemm<<<dim3(D_MODEL / 128, ROWS / 128, 3), 256, 0, stream>>>(Xh, Wt, QKV);
  attn_fwd<<<dim3(SEQ / 64, BATCH * N_HEADS), 128, 0, stream>>>(Qh, Kh, Vh, out);
}
